// GroupedQueryAttn_63608465653788
// MI455X (gfx1250) — compile-verified, matrix operand order checked by reading, not run on hardware
//
#include <hip/hip_runtime.h>

// ---------------------------------------------------------------------------
// GQA forward for MI455X (gfx1250): all matmuls on v_wmma_f32_16x16x32_bf16.
// Working set (~48MB bf16) is L2-resident (192MB L2), so GEMM fragments are
// loaded directly from global; LDS is used only for the P-tile lane transpose
// inside flash-attention.
//   - GEMM wave tile: 64x32 (4x2 accumulators), k-loop unrolled 2x so the
//     scheduler double-buffers fragment regs (avoids WMMA->VALU WAR nops)
//   - Attention wave: 32 Q rows x full T stream, K/V frags shared by both
//     16-row blocks -> 16 wmma : 8 fragment loads per 32-key chunk
//   - Softmax in base-2: v_exp_f32 is natively exp2, fold log2(e) into scale
// ---------------------------------------------------------------------------

#define MODEL_DIM 2048
#define S_LEN     2048
#define KV_DIM    512
#define QKV_N     (MODEL_DIM + 2 * KV_DIM)   // 3072
#define N_GROUPS  8
#define HEAD_DIM  64
#define N_HEADS   32
#define SM_SCALE2 0.1803368801111204f        // log2(e)/sqrt(64)

typedef __attribute__((ext_vector_type(16))) __bf16 v16bf;
typedef __attribute__((ext_vector_type(8)))  float  v8f;

// ---------------- fp32 -> bf16 cast ----------------
__global__ void cast_f32_bf16(const float* __restrict__ src,
                              __bf16* __restrict__ dst, int n) {
  int i = blockIdx.x * blockDim.x + threadIdx.x;
  if (i < n) dst[i] = (__bf16)src[i];
}

// ---------------- WMMA fragment helpers ----------------
// A fragment (16x32, MxK) per ISA 7.12.2: lane m=(lane&15), h=(lane>>4);
// elements 0..7  <- K = h*8 + 0..7
// elements 8..15 <- K = 16 + h*8 + 0..7
// B fragment for D = A*B has the identical layout applied to rows of B^T,
// i.e. load row-major (N x K) data exactly like an A fragment.
__device__ __forceinline__ v16bf load_frag(const __bf16* __restrict__ base,
                                           int ld, int row0, int k0) {
  const int lane = threadIdx.x & 31;
  const int r = row0 + (lane & 15);
  const int h = lane >> 4;
  const __bf16* p = base + (size_t)r * ld + k0 + h * 8;
  union { v16bf v; uint4 u[2]; } t;
  t.u[0] = *(const uint4*)(p);        // K = h*8   .. h*8+7   (16B)
  t.u[1] = *(const uint4*)(p + 16);   // K = 16+h*8 .. +7     (16B)
  return t.v;
}

__device__ __forceinline__ v8f wmma_bf16(v16bf a, v16bf b, v8f c) {
  return __builtin_amdgcn_wmma_f32_16x16x32_bf16(
      /*neg_a=*/false, a, /*neg_b=*/false, b,
      /*c_mod=*/(short)0, c, /*reuse_a=*/false, /*reuse_b=*/false);
}

// Shared 64x32 wave-tile GEMM core: acc[4][2] over rows row0+16i, cols col0+16j.
// Unroll 2x: two independent fragment register sets per iteration hide the
// WMMA->load WAR hazard and amortize address arithmetic over 16 WMMAs.
__device__ __forceinline__ void gemm_tile_64x32(const __bf16* __restrict__ A,
                                                const __bf16* __restrict__ B,
                                                int row0, int col0,
                                                v8f acc[4][2]) {
#pragma unroll 2
  for (int k0 = 0; k0 < MODEL_DIM; k0 += 32) {
    v16bf af[4], bf[2];
#pragma unroll
    for (int i = 0; i < 4; ++i) af[i] = load_frag(A, MODEL_DIM, row0 + 16 * i, k0);
#pragma unroll
    for (int j = 0; j < 2; ++j) bf[j] = load_frag(B, MODEL_DIM, col0 + 16 * j, k0);
#pragma unroll
    for (int i = 0; i < 4; ++i)
#pragma unroll
      for (int j = 0; j < 2; ++j) acc[i][j] = wmma_bf16(af[i], bf[j], acc[i][j]);
  }
}

// ---------------- Kernel 2: fused QKV GEMM ----------------
// C[s][n] = sum_k X[s][k] * Wqkv[n][k] + b[n]   (M=2048, N=3072, K=2048)
// Wave tile 64x32. Epilogue scatters:
//   n <  2048          -> Q[s][n]                 (row-major)
//   2048 <= n < 2560   -> K[g][t=s][d]            (d contiguous)
//   2560 <= n          -> V^T[g][d][t=s]          (t contiguous)
__global__ void __launch_bounds__(256)
gemm_qkv(const __bf16* __restrict__ X, const __bf16* __restrict__ W,
         const float* __restrict__ bias,
         __bf16* __restrict__ Q, __bf16* __restrict__ Kd,
         __bf16* __restrict__ VT) {
  const int wave = threadIdx.x >> 5;
  const int lane = threadIdx.x & 31;
  const int tile = blockIdx.x * 8 + wave;
  const int TN = QKV_N / 32;                 // 96 col tiles
  const int tm = tile / TN, tn = tile % TN;
  if (tm >= S_LEN / 64) return;
  const int row0 = tm * 64, col0 = tn * 32;

  v8f acc[4][2] = {};
  gemm_tile_64x32(X, W, row0, col0, acc);

  const int hs = lane >> 4;
#pragma unroll
  for (int i = 0; i < 4; ++i)
#pragma unroll
    for (int j = 0; j < 2; ++j)
#pragma unroll
      for (int r = 0; r < 8; ++r) {
        const int row = row0 + i * 16 + r + 8 * hs;
        const int col = col0 + j * 16 + (lane & 15);
        const __bf16 bv = (__bf16)(acc[i][j][r] + bias[col]);
        if (col < MODEL_DIM) {
          Q[(size_t)row * MODEL_DIM + col] = bv;
        } else if (col < MODEL_DIM + KV_DIM) {
          const int cc = col - MODEL_DIM, g = cc >> 6, d = cc & 63;
          Kd[((size_t)g * S_LEN + row) * HEAD_DIM + d] = bv;
        } else {
          const int cc = col - MODEL_DIM - KV_DIM, g = cc >> 6, d = cc & 63;
          VT[((size_t)g * HEAD_DIM + d) * S_LEN + row] = bv;
        }
      }
}

// ---------------- Kernel 3: flash attention ----------------
// One wave per (head, 32-row Q block); streams over T in chunks of 32 keys.
// Per chunk: 4 K-frag + 4 V-frag loads shared by both 16-row blocks,
// 8 S-WMMA + 8 PV-WMMA. Online softmax uses shfl_xor reductions within each
// 16-lane half (matching the C-fragment row split m / m+8); P is re-laid into
// an A fragment through LDS (no direct register path for that transpose).
__global__ void __launch_bounds__(128)
attn_kernel(const __bf16* __restrict__ Q, const __bf16* __restrict__ Kd,
            const __bf16* __restrict__ VT, __bf16* __restrict__ O) {
  __shared__ float pS[4][2][16][32];         // per-wave, per row-block (16KB)
  const int wave = threadIdx.x >> 5;
  const int lane = threadIdx.x & 31;
  const int task = blockIdx.x * 4 + wave;    // 0 .. 2047
  const int head = task >> 6;                // 0 .. 31  (h = g*4 + r)
  const int qblk = task & 63;                // 0 .. 63  (32 rows each)
  const int g    = head >> 2;
  const int qcol = head * HEAD_DIM;          // g*256 + r*64
  const int s0   = qblk * 32;
  const int hs   = lane >> 4;

  const __bf16* Kg = Kd + (size_t)g * S_LEN * HEAD_DIM;   // [t][d]
  const __bf16* Vg = VT + (size_t)g * HEAD_DIM * S_LEN;   // [d][t]

  v16bf qf[2][2];
#pragma unroll
  for (int rb = 0; rb < 2; ++rb) {
    qf[rb][0] = load_frag(Q, MODEL_DIM, s0 + rb * 16, qcol);        // d = 0..31
    qf[rb][1] = load_frag(Q, MODEL_DIM, s0 + rb * 16, qcol + 32);   // d = 32..63
  }

  float m_run[2][8], l_run[2][8];
#pragma unroll
  for (int rb = 0; rb < 2; ++rb)
    for (int j = 0; j < 8; ++j) { m_run[rb][j] = -1e30f; l_run[rb][j] = 0.f; }
  v8f o_acc[2][4] = {};

  for (int t0 = 0; t0 < S_LEN; t0 += 32) {
    // K/V fragments for this 32-key chunk, shared by both row blocks.
    const v16bf k00 = load_frag(Kg, HEAD_DIM, t0,      0);
    const v16bf k01 = load_frag(Kg, HEAD_DIM, t0,      32);
    const v16bf k10 = load_frag(Kg, HEAD_DIM, t0 + 16, 0);
    const v16bf k11 = load_frag(Kg, HEAD_DIM, t0 + 16, 32);
    v16bf vf[4];
#pragma unroll
    for (int c = 0; c < 4; ++c) vf[c] = load_frag(Vg, S_LEN, c * 16, t0);

#pragma unroll
    for (int rb = 0; rb < 2; ++rb) {
      // --- S tiles: 16 rows x 32 key positions ---
      v8f sa = {}, sb = {};
      sa = wmma_bf16(qf[rb][0], k00, sa);
      sa = wmma_bf16(qf[rb][1], k01, sa);
      sb = wmma_bf16(qf[rb][0], k10, sb);
      sb = wmma_bf16(qf[rb][1], k11, sb);

      // --- online softmax, base-2, in C-layout (row j+8*hs in vgpr j) ---
      float alpha[8], p0[8], p1[8];
#pragma unroll
      for (int j = 0; j < 8; ++j) {
        const float a = sa[j] * SM_SCALE2;   // already in log2 domain
        const float b = sb[j] * SM_SCALE2;
        float mx = fmaxf(a, b);
        for (int off = 1; off < 16; off <<= 1) mx = fmaxf(mx, __shfl_xor(mx, off));
        const float m_new = fmaxf(m_run[rb][j], mx);
        const float al = exp2f(m_run[rb][j] - m_new);
        const float e0 = exp2f(a - m_new);
        const float e1 = exp2f(b - m_new);
        float rs = e0 + e1;
        for (int off = 1; off < 16; off <<= 1) rs += __shfl_xor(rs, off);
        l_run[rb][j] = l_run[rb][j] * al + rs;
        m_run[rb][j] = m_new;
        alpha[j] = al; p0[j] = e0; p1[j] = e1;
      }
#pragma unroll
      for (int c = 0; c < 4; ++c)
#pragma unroll
        for (int j = 0; j < 8; ++j) o_acc[rb][c][j] *= alpha[j];

      // --- C-layout -> A-fragment transpose of P via LDS ---
      float (*ps)[32] = pS[wave][rb];
#pragma unroll
      for (int j = 0; j < 8; ++j) {
        const int m = j + 8 * hs;
        ps[m][lane & 15]        = p0[j];   // local t = 0..15
        ps[m][16 + (lane & 15)] = p1[j];   // local t = 16..31
      }
      union { v16bf v; __bf16 e[16]; } pf;
      {
        const int m = lane & 15, kb = hs * 8;
#pragma unroll
        for (int ii = 0; ii < 8; ++ii) {
          pf.e[ii]     = (__bf16)ps[m][kb + ii];
          pf.e[8 + ii] = (__bf16)ps[m][16 + kb + ii];
        }
      }

      // --- O += P @ V ; B^T rows are V^T[d][t], row-major in t ---
#pragma unroll
      for (int c = 0; c < 4; ++c)
        o_acc[rb][c] = wmma_bf16(pf.v, vf[c], o_acc[rb][c]);
    }
  }

  // --- normalize and store (attn-out already in fused flat layout) ---
#pragma unroll
  for (int rb = 0; rb < 2; ++rb)
#pragma unroll
    for (int c = 0; c < 4; ++c)
#pragma unroll
      for (int j = 0; j < 8; ++j) {
        const int row = s0 + rb * 16 + j + 8 * hs;
        const int col = qcol + c * 16 + (lane & 15);
        O[(size_t)row * MODEL_DIM + col] = (__bf16)(o_acc[rb][c][j] / l_run[rb][j]);
      }
}

// ---------------- Kernel 4: output projection ----------------
// out[s][n] = sum_k AO[s][k] * Wout[n][k] + b_out[n]   (fp32 result)
__global__ void __launch_bounds__(256)
gemm_out(const __bf16* __restrict__ A, const __bf16* __restrict__ W,
         const float* __restrict__ bias, float* __restrict__ out) {
  const int wave = threadIdx.x >> 5;
  const int lane = threadIdx.x & 31;
  const int tile = blockIdx.x * 8 + wave;
  const int TN = MODEL_DIM / 32;             // 64 col tiles
  const int tm = tile / TN, tn = tile % TN;
  if (tm >= S_LEN / 64) return;
  const int row0 = tm * 64, col0 = tn * 32;

  v8f acc[4][2] = {};
  gemm_tile_64x32(A, W, row0, col0, acc);

  const int hs = lane >> 4;
#pragma unroll
  for (int i = 0; i < 4; ++i)
#pragma unroll
    for (int j = 0; j < 2; ++j)
#pragma unroll
      for (int r = 0; r < 8; ++r) {
        const int row = row0 + i * 16 + r + 8 * hs;
        const int col = col0 + j * 16 + (lane & 15);
        out[(size_t)row * MODEL_DIM + col] = acc[i][j][r] + bias[col];
      }
}

// ---------------- host launcher ----------------
extern "C" void kernel_launch(void* const* d_in, const int* in_sizes, int n_in,
                              void* d_out, int out_size, void* d_ws, size_t ws_size,
                              hipStream_t stream) {
  const float* x     = (const float*)d_in[0];
  const float* w_qkv = (const float*)d_in[1];
  const float* b_qkv = (const float*)d_in[2];
  const float* w_out = (const float*)d_in[3];
  const float* b_out = (const float*)d_in[4];
  float* out = (float*)d_out;

  // Workspace carve-out (all 256B-aligned by construction), total ~48 MiB.
  char* ws = (char*)d_ws;
  size_t off = 0;
  __bf16* xb    = (__bf16*)(ws + off); off += (size_t)S_LEN * MODEL_DIM * 2;
  __bf16* wqkvb = (__bf16*)(ws + off); off += (size_t)QKV_N * MODEL_DIM * 2;
  __bf16* woutb = (__bf16*)(ws + off); off += (size_t)MODEL_DIM * MODEL_DIM * 2;
  __bf16* qb    = (__bf16*)(ws + off); off += (size_t)S_LEN * MODEL_DIM * 2;
  __bf16* kb    = (__bf16*)(ws + off); off += (size_t)N_GROUPS * S_LEN * HEAD_DIM * 2;
  __bf16* vtb   = (__bf16*)(ws + off); off += (size_t)N_GROUPS * HEAD_DIM * S_LEN * 2;
  __bf16* aob   = (__bf16*)(ws + off); off += (size_t)S_LEN * MODEL_DIM * 2;

  const int nx = S_LEN * MODEL_DIM;          // 4M
  const int nw = QKV_N * MODEL_DIM;          // 6M
  cast_f32_bf16<<<(nx + 255) / 256, 256, 0, stream>>>(x,     xb,    nx);
  cast_f32_bf16<<<(nw + 255) / 256, 256, 0, stream>>>(w_qkv, wqkvb, nw);
  cast_f32_bf16<<<(nx + 255) / 256, 256, 0, stream>>>(w_out, woutb, nx);

  // QKV GEMM: (2048/64)*(3072/32) = 3072 wave tiles, 8 waves/block.
  gemm_qkv<<<384, 256, 0, stream>>>(xb, wqkvb, b_qkv, qb, kb, vtb);

  // Attention: 32 heads * 64 q-blocks(32 rows) = 2048 wave tasks, 4 waves/block.
  attn_kernel<<<512, 128, 0, stream>>>(qb, kb, vtb, aob);

  // Output projection: (2048/64)*(2048/32) = 2048 wave tiles, 8 waves/block.
  gemm_out<<<256, 256, 0, stream>>>(aob, woutb, b_out, out);
}
